// HWNNLayer_42442866819263
// MI455X (gfx1250) — matrix operand order repaired and checked by reading, not compiled, not run on hardware
//
#include <hip/hip_runtime.h>
#include <hip/hip_bf16.h>

// MI455X (gfx1250) implementation.
// Roofline: streams wavelets + wavelets_inv (2 x 1 GiB fp32) once each from
// HBM -> ~92 us floor at 23.3 TB/s. Compute needed to stay BW-bound is only
// ~373 TFLOP/s, far under the v_wmma_f32_16x16x32_bf16 ceiling, so fp32 is
// converted to bf16 hi/lo pairs in registers and each tile does 3 WMMAs
// (Ah*Bh + Ah*Bl + Al*Bh) for near-fp32 accuracy at zero extra bandwidth.
// K is split 4-way across waves of a workgroup (deterministic LDS reduction)
// to raise occupancy from 1024 to 4096 waves so in-flight loads exceed the
// ~23 MB bandwidth-delay product of HBM.

#define NN 16384
#define NF 32
#define KSPLIT 4
#define KCHUNK (NN / KSPLIT)   // 4096

typedef __attribute__((ext_vector_type(16))) __bf16 v16bf;
typedef __attribute__((ext_vector_type(8)))  __bf16 v8bf;
typedef __attribute__((ext_vector_type(8)))  float  v8f;

__device__ __forceinline__ __bf16 bf_hi(float f) { return (__bf16)f; }
__device__ __forceinline__ __bf16 bf_lo(float f, __bf16 h) {
    return (__bf16)(f - (float)h);
}

// ---------------------------------------------------------------------------
// Kernel 1: x = features @ W, stored transposed [32, N] as bf16 hi/lo so the
// big GEMMs can load WMMA B-tiles as contiguous per-lane 32B chunks.
// ---------------------------------------------------------------------------
__global__ __launch_bounds__(256) void prep_x_kernel(
    const float* __restrict__ feat,   // [N, 32]
    const float* __restrict__ W,      // [32, 32]
    __bf16* __restrict__ xtHi,        // [32, N]
    __bf16* __restrict__ xtLo)        // [32, N]
{
    int tid = blockIdx.x * 256 + threadIdx.x;
    int col = tid & 31;
    int row = tid >> 5;
    if (row >= NN) return;
    float s = 0.0f;
    #pragma unroll
    for (int k = 0; k < NF; ++k)
        s = fmaf(feat[row * NF + k], W[k * NF + col], s);
    __bf16 h = bf_hi(s);
    xtHi[(size_t)col * NN + row] = h;
    xtLo[(size_t)col * NN + row] = bf_lo(s, h);
}

// ---------------------------------------------------------------------------
// Kernel 2/3: out[N,32] = A[N,N] @ B[N,32]  (B given transposed, bf16 hi/lo).
// Block = 256 threads = 8 waves = 2 row-blocks x 4 K-split waves.
// Each wave: 16 rows x 4096 K with v_wmma_f32_16x16x32_bf16 (2 N-tiles,
// 3 WMMAs per tile for the hi/lo split). Partials combined via LDS in a
// fixed order. Optional fused diag row-scale; output either fp32 row-major
// (final) or transposed bf16 hi/lo (intermediate y).
// ---------------------------------------------------------------------------
template <bool APPLY_DIAG, bool STORE_F32>
__global__ __launch_bounds__(256) void skinny_gemm_kernel(
    const float*  __restrict__ A,      // [N, N] row-major fp32
    const __bf16* __restrict__ Bhi,    // [32, N]
    const __bf16* __restrict__ Blo,    // [32, N]
    const float*  __restrict__ diag,   // [N] (only if APPLY_DIAG)
    float*        __restrict__ outF,   // [N, 32] (only if STORE_F32)
    __bf16*       __restrict__ outHiT, // [32, N] (only if !STORE_F32)
    __bf16*       __restrict__ outLoT) // [32, N]
{
    __shared__ float lds_acc[8 * 512];   // per wave: 2 tiles * 8 * 32 floats

    const int lane = threadIdx.x & 31;
    const int wib  = threadIdx.x >> 5;   // wave in block: 0..7
    const int wg   = wib >> 2;           // row-block within block: 0..1
    const int ks   = wib & 3;            // K-split id: 0..3
    const int r0   = (blockIdx.x * 2 + wg) * 16;  // this group's 16 rows
    const int row  = lane & 15;          // A row within tile / B column
    const int half = lane >> 4;          // 0 or 1

    // A operand, contiguous k order chosen here (same as B): lanes 0-15 hold K 0..15 of row=lane,
    // lanes 16-31 hold K 16..31 of row=lane-16.
    const float* Aptr = A + (size_t)(r0 + row) * NN + (size_t)half * 16;

    // B operand, same contiguous k order as A: lane n -> column n; lanes 0-15 hold K 0..15,
    // lanes 16-31 hold K 16..31. Transposed storage makes each lane's 16
    // values contiguous (32B, 32B-aligned).
    const size_t kbB = (size_t)half * 16;
    const __bf16* B0h = Bhi + (size_t)row * NN + kbB;
    const __bf16* B0l = Blo + (size_t)row * NN + kbB;
    const __bf16* B1h = Bhi + (size_t)(row + 16) * NN + kbB;
    const __bf16* B1l = Blo + (size_t)(row + 16) * NN + kbB;

    v8f acc0 = {};
    v8f acc1 = {};

    const int k_begin = ks * KCHUNK;
    const int k_end   = k_begin + KCHUNK;

    #pragma unroll 2
    for (int k0 = k_begin; k0 < k_end; k0 += 32) {
        // --- load 16 fp32 of A for this lane (two 32B chunks) ---
        float4 a0 = *(const float4*)(Aptr + k0);
        float4 a1 = *(const float4*)(Aptr + k0 + 4);
        float4 a2 = *(const float4*)(Aptr + k0 + 8);
        float4 a3 = *(const float4*)(Aptr + k0 + 12);

        // --- load B tiles (bf16, contiguous per lane, L2-resident) ---
        v16bf b0h = *(const v16bf*)(B0h + k0);
        v16bf b0l = *(const v16bf*)(B0l + k0);
        v16bf b1h = *(const v16bf*)(B1h + k0);
        v16bf b1l = *(const v16bf*)(B1l + k0);

        // --- split A into bf16 hi/lo ---
        const float av[16] = {a0.x, a0.y, a0.z, a0.w, a1.x, a1.y, a1.z, a1.w,
                              a2.x, a2.y, a2.z, a2.w, a3.x, a3.y, a3.z, a3.w};
        v16bf ah, al;
        #pragma unroll
        for (int i = 0; i < 16; ++i) {
            __bf16 h = bf_hi(av[i]);
            ah[i] = h;
            al[i] = bf_lo(av[i], h);
        }

        // --- 3-term hi/lo product per N-tile, fp32 accumulate ---
        acc0 = __builtin_amdgcn_wmma_f32_16x16x32_bf16(false, ah, false, b0h,
                                                       (short)0, acc0, false, false);
        acc0 = __builtin_amdgcn_wmma_f32_16x16x32_bf16(false, ah, false, b0l,
                                                       (short)0, acc0, false, false);
        acc0 = __builtin_amdgcn_wmma_f32_16x16x32_bf16(false, al, false, b0h,
                                                       (short)0, acc0, false, false);

        acc1 = __builtin_amdgcn_wmma_f32_16x16x32_bf16(false, ah, false, b1h,
                                                       (short)0, acc1, false, false);
        acc1 = __builtin_amdgcn_wmma_f32_16x16x32_bf16(false, ah, false, b1l,
                                                       (short)0, acc1, false, false);
        acc1 = __builtin_amdgcn_wmma_f32_16x16x32_bf16(false, al, false, b1h,
                                                       (short)0, acc1, false, false);
    }

    // --- deterministic cross-wave K reduction through LDS ---
    if (ks != 0) {
        float* dst = &lds_acc[wib * 512];
        #pragma unroll
        for (int v = 0; v < 8; ++v) {
            dst[v * 32 + lane]       = acc0[v];
            dst[256 + v * 32 + lane] = acc1[v];
        }
    }
    __syncthreads();
    if (ks != 0) return;

    #pragma unroll
    for (int p = 1; p < KSPLIT; ++p) {
        const float* src = &lds_acc[(wib + p) * 512];
        #pragma unroll
        for (int v = 0; v < 8; ++v) {
            acc0[v] += src[v * 32 + lane];
            acc1[v] += src[256 + v * 32 + lane];
        }
    }

    // C/D layout: VGPR v, lanes 0-15 -> (M=v, N=lane); lanes 16-31 -> (M=v+8).
    const int mbase = half * 8;

    if (APPLY_DIAG) {
        #pragma unroll
        for (int v = 0; v < 8; ++v) {
            float dv = diag[r0 + mbase + v];
            acc0[v] *= dv;
            acc1[v] *= dv;
        }
    }

    if (STORE_F32) {
        #pragma unroll
        for (int v = 0; v < 8; ++v) {
            outF[(size_t)(r0 + mbase + v) * NF + row]      = acc0[v];
            outF[(size_t)(r0 + mbase + v) * NF + row + 16] = acc1[v];
        }
    } else {
        // 8 consecutive row-indices per lane -> pack to one b128 store each.
        v8bf h0, l0, h1, l1;
        #pragma unroll
        for (int v = 0; v < 8; ++v) {
            __bf16 a = bf_hi(acc0[v]);
            h0[v] = a; l0[v] = bf_lo(acc0[v], a);
            __bf16 b = bf_hi(acc1[v]);
            h1[v] = b; l1[v] = bf_lo(acc1[v], b);
        }
        size_t o0 = (size_t)row * NN + (size_t)(r0 + mbase);
        size_t o1 = (size_t)(row + 16) * NN + (size_t)(r0 + mbase);
        *(v8bf*)(outHiT + o0) = h0;
        *(v8bf*)(outLoT + o0) = l0;
        *(v8bf*)(outHiT + o1) = h1;
        *(v8bf*)(outLoT + o1) = l1;
    }
}

extern "C" void kernel_launch(void* const* d_in, const int* in_sizes, int n_in,
                              void* d_out, int out_size, void* d_ws, size_t ws_size,
                              hipStream_t stream) {
    const float* features     = (const float*)d_in[0]; // [N, 32]
    const float* wavelets     = (const float*)d_in[1]; // [N, N]
    const float* wavelets_inv = (const float*)d_in[2]; // [N, N]
    const float* diag_filter  = (const float*)d_in[3]; // [N]
    const float* weight_mat   = (const float*)d_in[4]; // [32, 32]
    float* out = (float*)d_out;                        // [N, 32]

    // Workspace: 4 x [32, N] bf16 planes = 4 MB total.
    __bf16* xt_hi = (__bf16*)d_ws;
    __bf16* xt_lo = xt_hi + (size_t)NF * NN;
    __bf16* yt_hi = xt_lo + (size_t)NF * NN;
    __bf16* yt_lo = yt_hi + (size_t)NF * NN;

    // 1) x = features @ W  (transposed bf16 hi/lo)
    prep_x_kernel<<<(NN * NF) / 256, 256, 0, stream>>>(features, weight_mat,
                                                       xt_hi, xt_lo);

    // 2) y = diag .* (wavelets_inv @ x)   -> transposed bf16 hi/lo
    // 4096 waves: 512 blocks x 8 waves (2 row-blocks x 4 K-split waves).
    const int blocks = NN / 32;
    skinny_gemm_kernel<true, false><<<blocks, 256, 0, stream>>>(
        wavelets_inv, xt_hi, xt_lo, diag_filter, nullptr, yt_hi, yt_lo);

    // 3) out = wavelets @ y   (fp32 row-major)
    skinny_gemm_kernel<false, true><<<blocks, 256, 0, stream>>>(
        wavelets, yt_hi, yt_lo, nullptr, out, nullptr, nullptr);
}
